// SurfaceDiceCalgary_33698313404998
// MI455X (gfx1250) — compile-verified, matrix operand order checked by reading, not run on hardware
//
#include <hip/hip_runtime.h>
#include <hip/hip_bf16.h>

// Volume geometry (from reference: shape (1,1,128,128,96), z contiguous)
#define XD 128
#define YD 128
#define ZD 96
#define SX (YD * ZD)          // 12288, stride for x+-1
#define NVOX (XD * YD * ZD)   // 1572864
#define P2_BLOCKS 1024
#define P2_THREADS 256
#define P2_STRIDE (P2_BLOCKS * P2_THREADS)  // 262144; NVOX = 6 * P2_STRIDE exactly

typedef __attribute__((ext_vector_type(16))) _Float16 v16h;
typedef __attribute__((ext_vector_type(8)))  float    v8f;

// ---------------------------------------------------------------------------
// Wave-wide sum via CDNA5 WMMA: A has the lane's value at element 0 (rest 0),
// B is all-ones, so D[m][n] = rowsum_m for every column n.  Per the 16x16 f32
// C/D layout, lane l holds column (l%16), rows 0-7 (l<16) or 8-15 (l>=16) in
// its 8 VGPRs.  Hence  sum(lane l's 8 floats) + sum(lane l+16's 8 floats)
// == total over all 32 lanes.  Caller accumulates the returned value from
// lanes 0 and 16 only.  Requires EXEC == all ones (full, convergent wave).
// Values must be small integers (exact in f16): here counts <= 12.
// ---------------------------------------------------------------------------
__device__ __forceinline__ float wave_sum_wmma(float val) {
    v16h a;
#pragma unroll
    for (int i = 0; i < 16; ++i) a[i] = (_Float16)0.0f;
    a[0] = (_Float16)val;
    v16h b;
#pragma unroll
    for (int i = 0; i < 16; ++i) b[i] = (_Float16)1.0f;
    v8f c;
#pragma unroll
    for (int i = 0; i < 8; ++i) c[i] = 0.0f;
    // D = A x B + C  -> v_wmma_f32_16x16x32_f16
    c = __builtin_amdgcn_wmma_f32_16x16x32_f16(
        /*neg_a=*/false, a, /*neg_b=*/false, b,
        /*c_mod=*/(short)0, c, /*reuse_a=*/false, /*reuse_b=*/false);
    float s = c[0] + c[1] + c[2] + c[3] + c[4] + c[5] + c[6] + c[7];
    return s;
}

// ---------------------------------------------------------------------------
// Pass 1: fused threshold + 6-connected boundary extraction.
//   pred = sigmoid(net)>0.5  <=>  net>0 ;  gt = target!=0
//   boundary(v) = fg(v) && !(all 6 neighbors fg), OOB treated as background.
// Writes one byte per voxel per mask into workspace (lives in L2).
// ---------------------------------------------------------------------------
__global__ void __launch_bounds__(256)
boundary_kernel(const float* __restrict__ net, const int* __restrict__ tgt,
                unsigned char* __restrict__ bp, unsigned char* __restrict__ bg) {
    const int v = blockIdx.x * blockDim.x + threadIdx.x;
    if (v >= NVOX) return;
    const int x = v / SX;
    const int r = v - x * SX;
    const int y = r / ZD;
    const int z = r - y * ZD;

    const bool hxm = (x > 0), hxp = (x < XD - 1);
    const bool hym = (y > 0), hyp = (y < YD - 1);
    const bool hzm = (z > 0), hzp = (z < ZD - 1);

    // prediction mask boundary
    const bool pc = net[v] > 0.0f;
    unsigned char pb = 0;
    if (pc) {
        const bool pint = (hxm && net[v - SX] > 0.0f) && (hxp && net[v + SX] > 0.0f) &&
                          (hym && net[v - ZD] > 0.0f) && (hyp && net[v + ZD] > 0.0f) &&
                          (hzm && net[v - 1]  > 0.0f) && (hzp && net[v + 1]  > 0.0f);
        pb = pint ? 0 : 1;
    }
    bp[v] = pb;

    // ground-truth mask boundary
    const bool gc = tgt[v] != 0;
    unsigned char gb = 0;
    if (gc) {
        const bool gint = (hxm && tgt[v - SX] != 0) && (hxp && tgt[v + SX] != 0) &&
                          (hym && tgt[v - ZD] != 0) && (hyp && tgt[v + ZD] != 0) &&
                          (hzm && tgt[v - 1]  != 0) && (hzp && tgt[v + 1]  != 0);
        gb = gint ? 0 : 1;
    }
    bg[v] = gb;
}

// ---------------------------------------------------------------------------
// Pass 2: TOL=1 surface-overlap + reduction.
//   d2_to_other(v) <= 1  <=>  other-boundary at v or a 6-neighbor of v.
// Grid-stride (exact 6 iterations/thread), per-thread integer counters,
// WMMA wave reduction, LDS ds_add_f32 block combine, one global atomic per
// counter per block.
// ---------------------------------------------------------------------------
__global__ void __launch_bounds__(P2_THREADS)
overlap_kernel(const unsigned char* __restrict__ bp,
               const unsigned char* __restrict__ bg,
               float* __restrict__ acc) {
    const int tid = blockIdx.x * blockDim.x + threadIdx.x;
    float c_ovp = 0.0f, c_ovg = 0.0f, c_den = 0.0f;

    for (int v = tid; v < NVOX; v += P2_STRIDE) {
        const unsigned char pbc = bp[v];
        const unsigned char gbc = bg[v];
        if (pbc | gbc) {
            const int x = v / SX;
            const int r = v - x * SX;
            const int y = r / ZD;
            const int z = r - y * ZD;
            const bool hxm = (x > 0), hxp = (x < XD - 1);
            const bool hym = (y > 0), hyp = (y < YD - 1);
            const bool hzm = (z > 0), hzp = (z < ZD - 1);
            if (pbc) {
                const bool ok = gbc || (hxm && bg[v - SX]) || (hxp && bg[v + SX]) ||
                                       (hym && bg[v - ZD]) || (hyp && bg[v + ZD]) ||
                                       (hzm && bg[v - 1])  || (hzp && bg[v + 1]);
                if (ok) c_ovp += 1.0f;
                c_den += 1.0f;
            }
            if (gbc) {
                const bool ok = pbc || (hxm && bp[v - SX]) || (hxp && bp[v + SX]) ||
                                       (hym && bp[v - ZD]) || (hyp && bp[v + ZD]) ||
                                       (hzm && bp[v - 1])  || (hzp && bp[v + 1]);
                if (ok) c_ovg += 1.0f;
                c_den += 1.0f;
            }
        }
    }

    // Wave-level reduction through the matrix pipe (all waves full & convergent
    // here -> EXEC all ones).  Counts <= 12, exact in f16.
    const float s0 = wave_sum_wmma(c_ovp);
    const float s1 = wave_sum_wmma(c_ovg);
    const float s2 = wave_sum_wmma(c_den);

    __shared__ float bacc[3];
    if (threadIdx.x < 3) bacc[threadIdx.x] = 0.0f;
    __syncthreads();
    const int lane = threadIdx.x & 31;
    if (lane == 0 || lane == 16) {        // column-0 partners: together = wave total
        atomicAdd(&bacc[0], s0);
        atomicAdd(&bacc[1], s1);
        atomicAdd(&bacc[2], s2);
    }
    __syncthreads();
    if (threadIdx.x < 3) atomicAdd(&acc[threadIdx.x], bacc[threadIdx.x]);
}

__global__ void zero_acc_kernel(float* __restrict__ acc) {
    if (threadIdx.x < 4) acc[threadIdx.x] = 0.0f;
}

__global__ void finalize_kernel(const float* __restrict__ acc, float* __restrict__ out) {
    const float den = acc[2];
    out[0] = (den > 0.0f) ? (acc[0] + acc[1]) / den : 0.0f;
}

// ---------------------------------------------------------------------------
extern "C" void kernel_launch(void* const* d_in, const int* in_sizes, int n_in,
                              void* d_out, int out_size, void* d_ws, size_t ws_size,
                              hipStream_t stream) {
    const float* net = (const float*)d_in[0];   // net_out, f32, NVOX elems
    const int*   tgt = (const int*)d_in[1];     // target, i32, NVOX elems
    float* out = (float*)d_out;

    unsigned char* bp = (unsigned char*)d_ws;
    unsigned char* bg = bp + NVOX;
    float* acc = (float*)((char*)d_ws + 2 * (size_t)NVOX);  // 2*NVOX is 16B aligned

    zero_acc_kernel<<<1, 32, 0, stream>>>(acc);
    boundary_kernel<<<NVOX / 256, 256, 0, stream>>>(net, tgt, bp, bg);
    overlap_kernel<<<P2_BLOCKS, P2_THREADS, 0, stream>>>(bp, bg, acc);
    finalize_kernel<<<1, 1, 0, stream>>>(acc, out);
}
